// LSTM_51376398795261
// MI455X (gfx1250) — matrix operand order repaired and checked by reading, not compiled, not run on hardware
//
#include <hip/hip_runtime.h>
#include <hip/hip_bf16.h>

typedef __attribute__((ext_vector_type(16))) __bf16 v16bf;
typedef __attribute__((ext_vector_type(8)))  float  v8f;

#define B_SZ   16384
#define D_IN   1024
#define D_HID  1024
#define K_TOT  (D_IN + D_HID)   // 2048

// ---- A-operand (16x32, M x K) register layout, CDNA5 wave32 ----
//   lane&15 = M row
//   lanes 0-15 : elems 0-7 = K[0..7],  elems 8-15 = K[16..23]
//   lanes 16-31: elems 0-7 = K[8..15], elems 8-15 = K[24..31]
// `p` must point at  row_base + k_base + 8*(lane>=16).
__device__ __forceinline__ v16bf load_a_bf16(const float* __restrict__ p) {
    float4 a0 = *(const float4*)(p);
    float4 a1 = *(const float4*)(p + 4);
    float4 b0 = *(const float4*)(p + 16);
    float4 b1 = *(const float4*)(p + 20);
    v16bf r;
    r[0]  = (__bf16)a0.x; r[1]  = (__bf16)a0.y; r[2]  = (__bf16)a0.z; r[3]  = (__bf16)a0.w;
    r[4]  = (__bf16)a1.x; r[5]  = (__bf16)a1.y; r[6]  = (__bf16)a1.z; r[7]  = (__bf16)a1.w;
    r[8]  = (__bf16)b0.x; r[9]  = (__bf16)b0.y; r[10] = (__bf16)b0.z; r[11] = (__bf16)b0.w;
    r[12] = (__bf16)b1.x; r[13] = (__bf16)b1.y; r[14] = (__bf16)b1.z; r[15] = (__bf16)b1.w;
    return r;
}

// ---- B-operand (32x16, K x N) register layout, CDNA5 wave32 ----
//   lane&15 = N col; same K order as A operand
//   lanes 0-15 : elems 0-7 = K[0..7],  elems 8-15 = K[16..23]; lanes 16-31: K[8..15], K[24..31]
// `p` must point at  W_row(n) + k_base + 8*(lane>=16).
__device__ __forceinline__ v16bf load_b_bf16(const float* __restrict__ p) {
    float4 a0 = *(const float4*)(p);
    float4 a1 = *(const float4*)(p + 4);
    float4 b0 = *(const float4*)(p + 16);
    float4 b1 = *(const float4*)(p + 20);
    v16bf r;
    r[0]  = (__bf16)a0.x; r[1]  = (__bf16)a0.y; r[2]  = (__bf16)a0.z; r[3]  = (__bf16)a0.w;
    r[4]  = (__bf16)a1.x; r[5]  = (__bf16)a1.y; r[6]  = (__bf16)a1.z; r[7]  = (__bf16)a1.w;
    r[8]  = (__bf16)b0.x; r[9]  = (__bf16)b0.y; r[10] = (__bf16)b0.z; r[11] = (__bf16)b0.w;
    r[12] = (__bf16)b1.x; r[13] = (__bf16)b1.y; r[14] = (__bf16)b1.z; r[15] = (__bf16)b1.w;
    return r;
}

__device__ __forceinline__ float sigmoid_f(float x) {
    return 1.0f / (1.0f + __expf(-x));
}
__device__ __forceinline__ float tanh_f(float x) {
    float e = __expf(2.0f * x);
    return (e - 1.0f) / (e + 1.0f);
}

#define WMMA_BF16(A, Bm, C) \
    __builtin_amdgcn_wmma_f32_16x16x32_bf16(false, (A), false, (Bm), (short)0, (C), false, false)

// Block: 256 threads = 8 waves, 4 (M) x 2 (N).
// Each wave: 32x16 output tile per gate (2 M-subtiles of 16x16), 4 gates.
// Block tile: 128 (batch) x 32 (hidden).
__global__ __launch_bounds__(256)
void lstm_fused_wmma(const float* __restrict__ x,
                     const float* __restrict__ prev_h,
                     const float* __restrict__ prev_c,
                     const float* __restrict__ Wi, const float* __restrict__ bi,
                     const float* __restrict__ Wf, const float* __restrict__ bf,
                     const float* __restrict__ Wo, const float* __restrict__ bo,
                     const float* __restrict__ Wz, const float* __restrict__ bz,
                     float* __restrict__ out)
{
    const int lane  = threadIdx.x & 31;
    const int wave  = threadIdx.x >> 5;
    const int wm    = wave & 3;        // 0..3  M sub-block
    const int wn    = wave >> 2;       // 0..1  N sub-block

    const int m0    = blockIdx.x * 128 + wm * 32;  // batch-row base (wave)
    const int n0    = blockIdx.y * 32  + wn * 16;  // hidden-col base (wave)

    const int lrow   = lane & 15;            // M (A) or N (B) index within 16
    const int khalfA = (lane >> 4) * 8;      // A-operand K half-offset
    const int khalfB = (lane >> 4) * 8;      // B-operand K half-offset

    // Weight row pointers (W is [D_HID, K_TOT], K contiguous)
    const float* wri = Wi + (size_t)(n0 + lrow) * K_TOT + khalfB;
    const float* wrf = Wf + (size_t)(n0 + lrow) * K_TOT + khalfB;
    const float* wro = Wo + (size_t)(n0 + lrow) * K_TOT + khalfB;
    const float* wrz = Wz + (size_t)(n0 + lrow) * K_TOT + khalfB;

    v8f acc_i0 = {}; v8f acc_i1 = {};
    v8f acc_f0 = {}; v8f acc_f1 = {};
    v8f acc_o0 = {}; v8f acc_o1 = {};
    v8f acc_z0 = {}; v8f acc_z1 = {};

    for (int kb = 0; kb < K_TOT; kb += 32) {
        // concat(x, prev_h): each 32-wide K step lies entirely in one tensor
        const float* abase = (kb < D_IN) ? x : prev_h;
        const int    kk    = (kb < D_IN) ? kb : (kb - D_IN);

        const float* ap0 = abase + (size_t)(m0 + lrow)      * D_IN + kk + khalfA;
        const float* ap1 = abase + (size_t)(m0 + 16 + lrow) * D_IN + kk + khalfA;

        if (kb + 32 < K_TOT) {
            __builtin_prefetch(ap0 + 32, 0, 3);      // global_prefetch_b8
            __builtin_prefetch(wri + kb + 32, 0, 3);
        }

        v16bf a0  = load_a_bf16(ap0);
        v16bf a1  = load_a_bf16(ap1);
        v16bf b_i = load_b_bf16(wri + kb);
        v16bf b_f = load_b_bf16(wrf + kb);
        v16bf b_o = load_b_bf16(wro + kb);
        v16bf b_z = load_b_bf16(wrz + kb);

        acc_i0 = WMMA_BF16(a0, b_i, acc_i0);
        acc_i1 = WMMA_BF16(a1, b_i, acc_i1);
        acc_f0 = WMMA_BF16(a0, b_f, acc_f0);
        acc_f1 = WMMA_BF16(a1, b_f, acc_f1);
        acc_o0 = WMMA_BF16(a0, b_o, acc_o0);
        acc_o1 = WMMA_BF16(a1, b_o, acc_o1);
        acc_z0 = WMMA_BF16(a0, b_z, acc_z0);
        acc_z1 = WMMA_BF16(a1, b_z, acc_z1);
    }

    // ---- Epilogue: bias + activations + cell update ----
    // C/D layout: VGPR r, lane L -> N = L&15, M = r + 8*(L>=16)
    const int col  = n0 + (lane & 15);
    const int rofs = (lane >> 4) * 8;

    const float bbi = bi[col];
    const float bbf = bf[col];
    const float bbo = bo[col];
    const float bbz = bz[col];

    float* __restrict__ h_out = out;
    float* __restrict__ c_out = out + (size_t)B_SZ * D_HID;

#pragma unroll
    for (int r = 0; r < 8; ++r) {
        // subtile 0: rows m0 + r + rofs
        {
            const int row = m0 + r + rofs;
            const size_t idx = (size_t)row * D_HID + col;
            float ig = sigmoid_f(acc_i0[r] + bbi);
            float fg = sigmoid_f(acc_f0[r] + bbf);
            float og = sigmoid_f(acc_o0[r] + bbo);
            float zg = tanh_f  (acc_z0[r] + bbz);
            float c  = ig * zg + fg * prev_c[idx];
            h_out[idx] = og * tanh_f(c);
            c_out[idx] = c;
        }
        // subtile 1: rows m0 + 16 + r + rofs
        {
            const int row = m0 + 16 + r + rofs;
            const size_t idx = (size_t)row * D_HID + col;
            float ig = sigmoid_f(acc_i1[r] + bbi);
            float fg = sigmoid_f(acc_f1[r] + bbf);
            float og = sigmoid_f(acc_o1[r] + bbo);
            float zg = tanh_f  (acc_z1[r] + bbz);
            float c  = ig * zg + fg * prev_c[idx];
            h_out[idx] = og * tanh_f(c);
            c_out[idx] = c;
        }
    }
}

extern "C" void kernel_launch(void* const* d_in, const int* in_sizes, int n_in,
                              void* d_out, int out_size, void* d_ws, size_t ws_size,
                              hipStream_t stream) {
    (void)in_sizes; (void)n_in; (void)out_size; (void)d_ws; (void)ws_size;

    const float* x      = (const float*)d_in[0];
    const float* prev_h = (const float*)d_in[1];
    const float* prev_c = (const float*)d_in[2];
    const float* Wi     = (const float*)d_in[3];
    const float* bi     = (const float*)d_in[4];
    const float* Wf     = (const float*)d_in[5];
    const float* bf     = (const float*)d_in[6];
    const float* Wo     = (const float*)d_in[7];
    const float* bo     = (const float*)d_in[8];
    const float* Wz     = (const float*)d_in[9];
    const float* bz     = (const float*)d_in[10];
    float* out          = (float*)d_out;

    dim3 grid(B_SZ / 128, D_HID / 32);   // 128 x 32 = 4096 blocks
    dim3 block(256);
    lstm_fused_wmma<<<grid, block, 0, stream>>>(x, prev_h, prev_c,
                                                Wi, bi, Wf, bf, Wo, bo, Wz, bz,
                                                out);
}
